// LSTMEncode_21818433863848
// MI455X (gfx1250) — matrix operand order repaired and checked by reading, not compiled, not run on hardware
//
#include <hip/hip_runtime.h>
#include <math.h>

#define B_DIM   16384
#define IN_DIM  512
#define HID_DIM 1024
#define OUT_DIM 512
#define KTOT    (IN_DIM + HID_DIM)

typedef __attribute__((ext_vector_type(16))) __bf16 v16bf;
typedef __attribute__((ext_vector_type(8)))  __bf16 v8bf;
typedef __attribute__((ext_vector_type(8)))  float  v8f;
typedef __attribute__((ext_vector_type(4)))  float  v4f;

struct GateArgs {
    const float* W[4];
    const float* U[4];
    const float* b[4];
};

#define LOG2E 1.44269504088896340736f

// Branch-free activations on the raw HW transcendentals (v_exp_f32 / v_rcp_f32).
// sigmoid: x->+inf: exp2(-inf)=0 -> rcp(1)=1 ; x->-inf: exp2(+inf)=inf -> rcp(inf)=0.
__device__ __forceinline__ float fast_sigmoid(float x) {
    return __builtin_amdgcn_rcpf(1.0f + __builtin_amdgcn_exp2f(-x * LOG2E));
}
// tanh: x->+inf: rcp(inf)=0 -> 1 ; x->-inf: exp2->0 -> 1-2 = -1.
__device__ __forceinline__ float fast_tanh(float x) {
    return 1.0f - 2.0f * __builtin_amdgcn_rcpf(1.0f + __builtin_amdgcn_exp2f(2.0f * x * LOG2E));
}

// LDS strides in bf16 elements (multiples of 8 -> 16B-aligned vector accesses,
// padded vs 32 to break bank-conflict patterns)
#define LDA   40               // A tile row stride
#define LDBK  40               // B tile column (k-major) stride
#define BGATE (32 * LDBK)      // one gate's B tile footprint

// ---------------------------------------------------------------------------
// Kernel A: gates = x@[Wi|Wf|Wo|Wc] + h@[Ui|Uf|Uo|Uc] + b, fused with the LSTM
// pointwise step. Workgroup tile: 128 rows x 32 hidden cols (x 4 gates).
// 8 waves = 4 (wm) x 2 (wn); each wave owns TWO 16-row sub-tiles and the same
// 16-col range for all four gates -> 8 WMMAs per K-step per wave, B-fragments
// reused across the two M sub-tiles, and the pointwise math stays in registers.
// ---------------------------------------------------------------------------
__global__ __launch_bounds__(256)
void lstm_gates_kernel(const float* __restrict__ x,
                       const float* __restrict__ h,
                       const float* __restrict__ c_prev,
                       GateArgs ga,
                       float* __restrict__ h_new,
                       float* __restrict__ c_new) {
    __shared__ alignas(16) __bf16 As[128 * LDA];       // 10.0 KB
    __shared__ alignas(16) __bf16 Bs[4 * BGATE];       // 10.0 KB

    const int tid  = threadIdx.x;
    const int lane = tid & 31;
    const int wave = tid >> 5;
    const int wm   = wave >> 1;          // 0..3 -> M sub-tiles 2*wm, 2*wm+1
    const int wn   = wave & 1;           // 0..1  N sub-tile

    const int m0 = blockIdx.x * 128;
    const int n0 = blockIdx.y * 32;

    // staging work split
    const int a_row = tid >> 1;          // 0..127
    const int a_kq  = (tid & 1) * 16;    // 0 or 16
    const int b_g   = tid >> 6;          // 0..3 gate
    const int b_t   = tid & 63;
    const int b_n   = b_t >> 1;          // 0..31 column
    const int b_kh  = (b_t & 1) * 16;    // 0 or 16 (k half)

    v8f acc[2][4] = {};                  // [m sub-tile][gate] f32 accumulators

    const int half = (lane < 16) ? 0 : 1;
    const int lmod = lane & 15;

    for (int kk = 0; kk < KTOT; kk += 32) {
        // ---- stage A tile (128 x 32) fp32 -> bf16, row-major ----
        {
            const float* src;
            int k;
            if (kk < IN_DIM) { src = x + (size_t)(m0 + a_row) * IN_DIM;  k = kk; }
            else             { src = h + (size_t)(m0 + a_row) * HID_DIM; k = kk - IN_DIM; }
            const v4f* p = (const v4f*)(src + k + a_kq);
            v4f f0 = p[0], f1 = p[1], f2 = p[2], f3 = p[3];
            v8bf t0, t1;
            #pragma unroll
            for (int j = 0; j < 4; ++j) {
                t0[j] = (__bf16)f0[j]; t0[4 + j] = (__bf16)f1[j];
                t1[j] = (__bf16)f2[j]; t1[4 + j] = (__bf16)f3[j];
            }
            __bf16* dst = &As[a_row * LDA + a_kq];
            *(v8bf*)dst       = t0;
            *(v8bf*)(dst + 8) = t1;
        }
        // ---- stage B tiles, 4 gates, 32k x 32n, stored k-major per column ----
        {
            const float* src;
            int k;
            if (kk < IN_DIM) { src = ga.W[b_g]; k = kk; }
            else             { src = ga.U[b_g]; k = kk - IN_DIM; }
            const float* col = src + (size_t)(k + b_kh) * HID_DIM + (n0 + b_n);
            v16bf tmpb;
            #pragma unroll
            for (int j = 0; j < 16; ++j) tmpb[j] = (__bf16)col[(size_t)j * HID_DIM];
            __bf16* dst = &Bs[b_g * BGATE + b_n * LDBK + b_kh];
            v8bf* tp = (v8bf*)&tmpb;
            *(v8bf*)dst       = tp[0];
            *(v8bf*)(dst + 8) = tp[1];
        }
        __syncthreads();

        // ---- fragments + 8 WMMAs (2 M sub-tiles x 4 gates) ----
        v16bf afrag[2];
        #pragma unroll
        for (int q = 0; q < 2; ++q) {
            // 16-bit A layout: lanes<16 -> K 0..7 / 16..23 ; lanes>=16 -> K 8..15 / 24..31
            v8bf* ap = (v8bf*)&afrag[q];
            const __bf16* arow = &As[((wm * 2 + q) * 16 + lmod) * LDA];
            ap[0] = *(const v8bf*)(arow + half * 8);
            ap[1] = *(const v8bf*)(arow + half * 8 + 16);
        }
        #pragma unroll
        for (int g = 0; g < 4; ++g) {
            // 16-bit B layout: lane = n, lanes<16 -> K 0..7 / 16..23, lanes>=16 -> K 8..15 / 24..31
            v16bf bfrag;
            v8bf* bp = (v8bf*)&bfrag;
            const __bf16* bcol = &Bs[g * BGATE + (wn * 16 + lmod) * LDBK + half * 8];
            bp[0] = *(const v8bf*)(bcol);
            bp[1] = *(const v8bf*)(bcol + 16);
            #pragma unroll
            for (int q = 0; q < 2; ++q) {
                acc[q][g] = __builtin_amdgcn_wmma_f32_16x16x32_bf16(
                    false, afrag[q], false, bfrag, (short)0, acc[q][g], false, false);
            }
        }
        __syncthreads();
    }

    // ---- epilogue: bias + LSTM pointwise (branch-free), write h_new / c_new ----
    // C/D layout: lane = n (mod 16); VGPR r -> m = r + (lane<16 ? 0 : 8)
    const int col = n0 + wn * 16 + lmod;
    const float bi  = ga.b[0][col];
    const float bff = ga.b[1][col];
    const float bo  = ga.b[2][col];
    const float bc  = ga.b[3][col];
    #pragma unroll
    for (int q = 0; q < 2; ++q) {
        const int mbase = m0 + (wm * 2 + q) * 16 + half * 8;
        #pragma unroll
        for (int r = 0; r < 8; ++r) {
            const size_t idx = (size_t)(mbase + r) * HID_DIM + col;
            float i_t = fast_sigmoid(acc[q][0][r] + bi);
            float f_t = fast_sigmoid(acc[q][1][r] + bff);
            float o_t = fast_sigmoid(acc[q][2][r] + bo);
            float g_t = fast_tanh(acc[q][3][r] + bc);
            float cn  = f_t * c_prev[idx] + i_t * g_t;
            float hn  = o_t * fast_tanh(cn);
            c_new[idx] = cn;
            h_new[idx] = hn;
        }
    }
}

// ---------------------------------------------------------------------------
// Kernel B: out = h_new @ fc_w^T + fc_b.   fc_w is [OUT, HID] row-major, so a
// B-fragment (fixed n, two runs of 8 consecutive k) is CONTIGUOUS in global memory ->
// both fragments load straight from global, no LDS. Each wave computes one
// 16-row M tile x FOUR 16-col N tiles, reusing its A-fragment 4x per K-step.
// ---------------------------------------------------------------------------
__global__ __launch_bounds__(256)
void lstm_fc_kernel(const float* __restrict__ hn,
                    const float* __restrict__ fcw,
                    const float* __restrict__ fcb,
                    float* __restrict__ out) {
    const int lane = threadIdx.x & 31;
    const int wave = threadIdx.x >> 5;
    const int job  = blockIdx.x * 8 + wave;     // 8192 wave jobs
    const int mt   = job >> 3;                  // 0..1023
    const int ng   = job & 7;                   // N group of 4 tiles
    const int m0   = mt * 16;
    const int obase = ng * 64;
    const int half = (lane < 16) ? 0 : 1;
    const int lmod = lane & 15;

    v8f acc[4] = {};

    const float* arow_base = hn + (size_t)(m0 + lmod) * HID_DIM;

    for (int kk = 0; kk < HID_DIM; kk += 32) {
        v16bf afrag;
        {
            const float* a0 = arow_base + kk + half * 8;        // K 0..7 / 8..15
            const float* a1 = a0 + 16;                          // K 16..23 / 24..31
            v4f f0 = ((const v4f*)a0)[0], f1 = ((const v4f*)a0)[1];
            v4f f2 = ((const v4f*)a1)[0], f3 = ((const v4f*)a1)[1];
            #pragma unroll
            for (int j = 0; j < 4; ++j) {
                afrag[j]      = (__bf16)f0[j];
                afrag[4 + j]  = (__bf16)f1[j];
                afrag[8 + j]  = (__bf16)f2[j];
                afrag[12 + j] = (__bf16)f3[j];
            }
        }
        #pragma unroll
        for (int nt = 0; nt < 4; ++nt) {
            v16bf bfrag;
            const float* b0 = fcw + (size_t)(obase + nt * 16 + lmod) * HID_DIM
                                  + kk + half * 8;              // K 0..7 / 8..15 (+16)
            v4f g0 = ((const v4f*)b0)[0], g1 = ((const v4f*)b0)[1];
            v4f g2 = ((const v4f*)(b0 + 16))[0], g3 = ((const v4f*)(b0 + 16))[1];
            #pragma unroll
            for (int j = 0; j < 4; ++j) {
                bfrag[j]      = (__bf16)g0[j];
                bfrag[4 + j]  = (__bf16)g1[j];
                bfrag[8 + j]  = (__bf16)g2[j];
                bfrag[12 + j] = (__bf16)g3[j];
            }
            acc[nt] = __builtin_amdgcn_wmma_f32_16x16x32_bf16(
                false, afrag, false, bfrag, (short)0, acc[nt], false, false);
        }
    }

    #pragma unroll
    for (int nt = 0; nt < 4; ++nt) {
        const int o = obase + nt * 16 + lmod;
        const float bias = fcb[o];
        #pragma unroll
        for (int r = 0; r < 8; ++r) {
            out[(size_t)(m0 + half * 8 + r) * OUT_DIM + o] = acc[nt][r] + bias;
        }
    }
}

extern "C" void kernel_launch(void* const* d_in, const int* in_sizes, int n_in,
                              void* d_out, int out_size, void* d_ws, size_t ws_size,
                              hipStream_t stream) {
    (void)in_sizes; (void)n_in; (void)out_size; (void)d_ws; (void)ws_size;

    const float* x = (const float*)d_in[0];
    const float* h = (const float*)d_in[1];
    const float* c = (const float*)d_in[2];
    GateArgs ga;
    ga.W[0] = (const float*)d_in[3];  ga.U[0] = (const float*)d_in[4];  ga.b[0] = (const float*)d_in[5];
    ga.W[1] = (const float*)d_in[6];  ga.U[1] = (const float*)d_in[7];  ga.b[1] = (const float*)d_in[8];
    ga.W[2] = (const float*)d_in[9];  ga.U[2] = (const float*)d_in[10]; ga.b[2] = (const float*)d_in[11];
    ga.W[3] = (const float*)d_in[12]; ga.U[3] = (const float*)d_in[13]; ga.b[3] = (const float*)d_in[14];
    const float* fcw = (const float*)d_in[15];
    const float* fcb = (const float*)d_in[16];

    // d_out = [ out (B*OUT) | h_new (B*HID) | c_new (B*HID) ] fp32
    float* out_p  = (float*)d_out;
    float* hnew_p = out_p  + (size_t)B_DIM * OUT_DIM;
    float* cnew_p = hnew_p + (size_t)B_DIM * HID_DIM;

    dim3 gridA(B_DIM / 128, HID_DIM / 32);
    lstm_gates_kernel<<<gridA, 256, 0, stream>>>(x, h, c, ga, hnew_p, cnew_p);

    dim3 gridB(((B_DIM / 16) * (OUT_DIM / 16)) / (8 * 4));
    lstm_fc_kernel<<<gridB, 256, 0, stream>>>(hnew_p, fcw, fcb, out_p);
}
